// SelfAttention_V1_with_V2_weights_2224793059597
// MI455X (gfx1250) — matrix operand order repaired and checked by reading, not compiled, not run on hardware
//
#include <hip/hip_runtime.h>

#define SEQ 4096
#define DIM 1024
#define LDS_STR 40   // padded row stride (elements) for Nx32 bf16 tiles; 80B keeps 16B align
                     // == TDM pad: 16 DWORDs data + 4 DWORDs pad per row

typedef __bf16 bf16;
typedef __attribute__((ext_vector_type(16))) __bf16 v16bf;
typedef __attribute__((ext_vector_type(8)))  float  v8f;
typedef __attribute__((ext_vector_type(4)))  unsigned int u32x4;
typedef __attribute__((ext_vector_type(8)))  int i32x8;
typedef __attribute__((ext_vector_type(4)))  int i32x4;

// ---------- helpers ----------

static __device__ __forceinline__ unsigned int pack2(float a, float b) {
    unsigned short ua = __builtin_bit_cast(unsigned short, (bf16)a);
    unsigned short ub = __builtin_bit_cast(unsigned short, (bf16)b);
    return ((unsigned int)ub << 16) | (unsigned int)ua;
}

// A fragment (16x32 bf16, MxK). Lane L<16 -> row M=L, k-halves {0..7,16..23};
// lane L>=16 -> row M=L-16, k-halves {8..15,24..31}. (ISA 7.12.2)
static __device__ __forceinline__ v16bf frag_a_from_lds(const bf16* tile, int row, int lane) {
    const int hi = lane >> 4;
    union { v16bf v; uint4 q[2]; } u;
    const uint4* p = (const uint4*)(tile + row * LDS_STR);
    u.q[0] = p[hi];
    u.q[1] = p[2 + hi];
    return u.v;
}

// B fragment (32x16 bf16, KxN) from an LDS tile stored as [n][k].
static __device__ __forceinline__ v16bf frag_b_from_lds(const bf16* tile, int col, int lane) {
    const int hi = lane >> 4;
    union { v16bf v; uint4 q[2]; } u;
    const uint4* p = (const uint4*)(tile + col * LDS_STR);
    u.q[0] = p[hi];
    u.q[1] = p[2 + hi];
    return u.v;
}

static __device__ __forceinline__ v8f wmma_bf16(v16bf a, v16bf b, v8f c) {
    return __builtin_amdgcn_wmma_f32_16x16x32_bf16(false, a, false, b, (short)0, c,
                                                   false, false);
}

// ---------- TDM: DMA a 2D bf16 tile (rows x 32 elems) global -> LDS ----------
// Descriptor packed per CDNA5 ISA. LDS padding:
// after every 16 DWORDs (one 64B row), skip 4 DWORDs -> LDS row stride 80B
// = LDS_STR elements, matching the fragment loaders above.
static __device__ __forceinline__ void tdm_load_tile(
        unsigned lds_byte_off,   // LDS destination byte offset
        const void* gptr,        // global address of tile element (0,0)
        unsigned rows,           // tile_dim1
        unsigned ld_elems)       // row pitch in elements (tensor_dim0_stride)
{
    const unsigned long long ga = (unsigned long long)gptr;
    const unsigned cols = 32u;   // tile_dim0 (contiguous bf16)

    u32x4 g0;
    g0[0] = 1u;                                            // count=1, user desc
    g0[1] = lds_byte_off;                                  // lds_addr
    g0[2] = (unsigned)(ga & 0xFFFFFFFFu);                  // global_addr[31:0]
    g0[3] = (unsigned)((ga >> 32) & 0x01FFFFFFu)           // global_addr[56:32]
          | 0x80000000u;                                   // type=2 ("image")

    i32x8 g1;
    g1[0] = (int)((1u << 16)                               // data_size = 2 bytes
                | (1u << 20)                               // pad_enable
                | (3u << 22)                               // pad_interval: 16 DWORDs
                | (3u << 25));                             // pad_amount: 4 DWORDs
    g1[1] = (int)((ld_elems & 0xFFFFu) << 16);             // tensor_dim0[15:0]
    g1[2] = (int)(((ld_elems >> 16) & 0xFFFFu)             // tensor_dim0[31:16]
                | ((rows & 0xFFFFu) << 16));               // tensor_dim1[15:0]
    g1[3] = (int)(((rows >> 16) & 0xFFFFu)                 // tensor_dim1[31:16]
                | (cols << 16));                           // tile_dim0
    g1[4] = (int)(rows & 0xFFFFu);                         // tile_dim1 (tile_dim2=0)
    g1[5] = (int)ld_elems;                                 // tensor_dim0_stride[31:0]
    g1[6] = 0;
    g1[7] = 0;

    const i32x4 z4 = {0, 0, 0, 0};
    const i32x8 z8 = {0, 0, 0, 0, 0, 0, 0, 0};
    __builtin_amdgcn_tensor_load_to_lds(g0, g1, z4, z4, z8, 0);
}

#define ZACC {0,0,0,0,0,0,0,0}

// ---------- kernel 1: QKV projection (fp32 x,W -> bf16 Q/K/V), 128x64 tile ----------

__global__ __launch_bounds__(128) void qkv_kernel(
        const float* __restrict__ x,
        const float* __restrict__ Wq, const float* __restrict__ Wk, const float* __restrict__ Wv,
        bf16* __restrict__ Qb, bf16* __restrict__ Kb, bf16* __restrict__ Vb) {
    __shared__ __align__(16) bf16 lx[128 * LDS_STR];
    __shared__ __align__(16) bf16 lw[64 * LDS_STR];

    const int tid  = threadIdx.x;
    const int lane = tid & 31;
    const int wave = tid >> 5;
    const int n0 = blockIdx.x * 64;
    const int m0 = blockIdx.y * 128;
    const float* W  = (blockIdx.z == 0) ? Wq : (blockIdx.z == 1) ? Wk : Wv;
    bf16*        Ot = (blockIdx.z == 0) ? Qb : (blockIdx.z == 1) ? Kb : Vb;

    v8f acc0[4] = { ZACC, ZACC, ZACC, ZACC };
    v8f acc1[4] = { ZACC, ZACC, ZACC, ZACC };

    for (int k0 = 0; k0 < DIM; k0 += 32) {
        // stage x tile: 128 rows x 32 k, one row per thread, fp32 -> bf16
        {
            const float4* src = (const float4*)(x + (size_t)(m0 + tid) * DIM + k0);
            uint4* dst = (uint4*)(lx + tid * LDS_STR);
#pragma unroll
            for (int h = 0; h < 2; ++h) {
                float4 f0 = src[4 * h + 0], f1 = src[4 * h + 1];
                float4 f2 = src[4 * h + 2], f3 = src[4 * h + 3];
                dst[2 * h + 0] = make_uint4(pack2(f0.x, f0.y), pack2(f0.z, f0.w),
                                            pack2(f1.x, f1.y), pack2(f1.z, f1.w));
                dst[2 * h + 1] = make_uint4(pack2(f2.x, f2.y), pack2(f2.z, f2.w),
                                            pack2(f3.x, f3.y), pack2(f3.z, f3.w));
            }
        }
        // stage W tile transposed: [n][k]
        for (int e = tid; e < 64 * 32; e += 128) {
            const int k = e >> 6, n = e & 63;
            lw[n * LDS_STR + k] = (bf16)W[(size_t)(k0 + k) * DIM + n0 + n];
        }
        __syncthreads();

        const v16bf a0 = frag_a_from_lds(lx, wave * 32 +      (lane & 15), lane);
        const v16bf a1 = frag_a_from_lds(lx, wave * 32 + 16 + (lane & 15), lane);
#pragma unroll
        for (int f = 0; f < 4; ++f) {
            const v16bf b = frag_b_from_lds(lw, f * 16 + (lane & 15), lane);
            acc0[f] = wmma_bf16(a0, b, acc0[f]);
            acc1[f] = wmma_bf16(a1, b, acc1[f]);
        }
        __syncthreads();
    }

    const int ncol = lane & 15;
    const int msub = 8 * (lane >> 4);
#pragma unroll
    for (int f = 0; f < 4; ++f)
#pragma unroll
        for (int r = 0; r < 8; ++r) {
            Ot[(size_t)(m0 + wave * 32 +      msub + r) * DIM + n0 + f * 16 + ncol] = (bf16)acc0[f][r];
            Ot[(size_t)(m0 + wave * 32 + 16 + msub + r) * DIM + n0 + f * 16 + ncol] = (bf16)acc1[f][r];
        }
}

// ---------- kernel 2: scores S = (Q @ K^T) / 32, TDM double-buffered, 128x64 tile ----------

__global__ __launch_bounds__(128) void scores_kernel(
        const bf16* __restrict__ Q, const bf16* __restrict__ K, bf16* __restrict__ S) {
    __shared__ __align__(16) bf16 lq[2][128 * LDS_STR];
    __shared__ __align__(16) bf16 lk[2][64 * LDS_STR];

    const int tid  = threadIdx.x;
    const int lane = tid & 31;
    const int wave = tid >> 5;
    const int n0 = blockIdx.x * 64;    // key tile
    const int m0 = blockIdx.y * 128;   // query tile

    const unsigned lq_off[2] = { (unsigned)(size_t)&lq[0][0], (unsigned)(size_t)&lq[1][0] };
    const unsigned lk_off[2] = { (unsigned)(size_t)&lk[0][0], (unsigned)(size_t)&lk[1][0] };

    v8f acc0[4] = { ZACC, ZACC, ZACC, ZACC };
    v8f acc1[4] = { ZACC, ZACC, ZACC, ZACC };

    const int NIT = DIM / 32;
    if (tid == 0) {
        tdm_load_tile(lq_off[0], Q + (size_t)m0 * DIM, 128, DIM);
        tdm_load_tile(lk_off[0], K + (size_t)n0 * DIM,  64, DIM);
    }

    for (int it = 0; it < NIT; ++it) {
        const int buf = it & 1;
        if (tid == 0) {
            if (it + 1 < NIT) {
                tdm_load_tile(lq_off[buf ^ 1], Q + (size_t)m0 * DIM + (it + 1) * 32, 128, DIM);
                tdm_load_tile(lk_off[buf ^ 1], K + (size_t)n0 * DIM + (it + 1) * 32,  64, DIM);
                __builtin_amdgcn_s_wait_tensorcnt(2);   // current pair done
            } else {
                __builtin_amdgcn_s_wait_tensorcnt(0);
            }
        }
        __syncthreads();

        const v16bf a0 = frag_a_from_lds(lq[buf], wave * 32 +      (lane & 15), lane);
        const v16bf a1 = frag_a_from_lds(lq[buf], wave * 32 + 16 + (lane & 15), lane);
#pragma unroll
        for (int f = 0; f < 4; ++f) {
            const v16bf b = frag_b_from_lds(lk[buf], f * 16 + (lane & 15), lane);
            acc0[f] = wmma_bf16(a0, b, acc0[f]);
            acc1[f] = wmma_bf16(a1, b, acc1[f]);
        }
        __syncthreads();   // readers done before buf re-fill
    }

    const float scale = 0.03125f;     // 1/sqrt(1024)
    const int ncol = lane & 15;
    const int msub = 8 * (lane >> 4);
#pragma unroll
    for (int f = 0; f < 4; ++f)
#pragma unroll
        for (int r = 0; r < 8; ++r) {
            S[(size_t)(m0 + wave * 32 +      msub + r) * SEQ + n0 + f * 16 + ncol] = (bf16)(acc0[f][r] * scale);
            S[(size_t)(m0 + wave * 32 + 16 + msub + r) * SEQ + n0 + f * 16 + ncol] = (bf16)(acc1[f][r] * scale);
        }
}

// ---------- kernel 3: row softmax in place (bf16 in / bf16 out) ----------

__global__ __launch_bounds__(256) void softmax_kernel(bf16* __restrict__ P) {
    __shared__ float red[256];
    const int tid = threadIdx.x;
    bf16* prow = P + (size_t)blockIdx.x * SEQ;

    float v[16];
    float mx = -3.0e38f;
#pragma unroll
    for (int i = 0; i < 16; ++i) {
        v[i] = (float)prow[tid + i * 256];
        mx = fmaxf(mx, v[i]);
    }
    red[tid] = mx;
    __syncthreads();
    for (int s = 128; s > 0; s >>= 1) {
        if (tid < s) red[tid] = fmaxf(red[tid], red[tid + s]);
        __syncthreads();
    }
    mx = red[0];
    __syncthreads();

    float sum = 0.0f;
#pragma unroll
    for (int i = 0; i < 16; ++i) {
        v[i] = __expf(v[i] - mx);
        sum += v[i];
    }
    red[tid] = sum;
    __syncthreads();
    for (int s = 128; s > 0; s >>= 1) {
        if (tid < s) red[tid] += red[tid + s];
        __syncthreads();
    }
    const float inv = 1.0f / red[0];
#pragma unroll
    for (int i = 0; i < 16; ++i)
        prow[tid + i * 256] = (bf16)(v[i] * inv);
}

// ---------- kernel 4: O = P @ V, TDM double-buffered P, 128x64 tile ----------

__global__ __launch_bounds__(128) void out_kernel(
        const bf16* __restrict__ P, const bf16* __restrict__ V, float* __restrict__ out) {
    __shared__ __align__(16) bf16 lp[2][128 * LDS_STR];
    __shared__ __align__(16) bf16 lv[64 * LDS_STR];

    const int tid  = threadIdx.x;
    const int lane = tid & 31;
    const int wave = tid >> 5;
    const int n0 = blockIdx.x * 64;
    const int m0 = blockIdx.y * 128;
    const unsigned lp_off[2] = { (unsigned)(size_t)&lp[0][0], (unsigned)(size_t)&lp[1][0] };

    v8f acc0[4] = { ZACC, ZACC, ZACC, ZACC };
    v8f acc1[4] = { ZACC, ZACC, ZACC, ZACC };

    const int NIT = SEQ / 32;
    if (tid == 0)
        tdm_load_tile(lp_off[0], P + (size_t)m0 * SEQ, 128, SEQ);

    for (int it = 0; it < NIT; ++it) {
        const int buf = it & 1;
        const int k0 = it * 32;

        // V tile transposed into LDS [n][k] (overlaps with the TDM transfer)
        for (int e = tid; e < 64 * 32; e += 128) {
            const int k = e >> 6, n = e & 63;
            lv[n * LDS_STR + k] = V[(size_t)(k0 + k) * DIM + n0 + n];
        }
        if (tid == 0) {
            if (it + 1 < NIT) {
                tdm_load_tile(lp_off[buf ^ 1], P + (size_t)m0 * SEQ + (it + 1) * 32, 128, SEQ);
                __builtin_amdgcn_s_wait_tensorcnt(1);   // current P tile done
            } else {
                __builtin_amdgcn_s_wait_tensorcnt(0);
            }
        }
        __syncthreads();

        const v16bf a0 = frag_a_from_lds(lp[buf], wave * 32 +      (lane & 15), lane);
        const v16bf a1 = frag_a_from_lds(lp[buf], wave * 32 + 16 + (lane & 15), lane);
#pragma unroll
        for (int f = 0; f < 4; ++f) {
            const v16bf b = frag_b_from_lds(lv, f * 16 + (lane & 15), lane);
            acc0[f] = wmma_bf16(a0, b, acc0[f]);
            acc1[f] = wmma_bf16(a1, b, acc1[f]);
        }
        __syncthreads();
    }

    const int ncol = lane & 15;
    const int msub = 8 * (lane >> 4);
#pragma unroll
    for (int f = 0; f < 4; ++f)
#pragma unroll
        for (int r = 0; r < 8; ++r) {
            out[(size_t)(m0 + wave * 32 +      msub + r) * DIM + n0 + f * 16 + ncol] = acc0[f][r];
            out[(size_t)(m0 + wave * 32 + 16 + msub + r) * DIM + n0 + f * 16 + ncol] = acc1[f][r];
        }
}

// ---------- launch ----------

extern "C" void kernel_launch(void* const* d_in, const int* in_sizes, int n_in,
                              void* d_out, int out_size, void* d_ws, size_t ws_size,
                              hipStream_t stream) {
    (void)in_sizes; (void)n_in; (void)out_size; (void)ws_size;

    const float* x  = (const float*)d_in[0];
    const float* Wq = (const float*)d_in[1];
    const float* Wk = (const float*)d_in[2];
    const float* Wv = (const float*)d_in[3];
    float* out = (float*)d_out;

    char* ws = (char*)d_ws;
    const size_t QKV_BYTES = (size_t)SEQ * DIM * sizeof(bf16);   // 8 MB each
    bf16* Qb = (bf16*)(ws + 0 * QKV_BYTES);
    bf16* Kb = (bf16*)(ws + 1 * QKV_BYTES);
    bf16* Vb = (bf16*)(ws + 2 * QKV_BYTES);
    bf16* Sb = (bf16*)(ws + 3 * QKV_BYTES);                      // 32 MB (SEQ x SEQ bf16)

    qkv_kernel   <<<dim3(DIM / 64, SEQ / 128, 3), 128, 0, stream>>>(x, Wq, Wk, Wv, Qb, Kb, Vb);
    scores_kernel<<<dim3(SEQ / 64, SEQ / 128),    128, 0, stream>>>(Qb, Kb, Sb);
    softmax_kernel<<<SEQ,                         256, 0, stream>>>(Sb);
    out_kernel   <<<dim3(DIM / 64, SEQ / 128),    128, 0, stream>>>(Sb, Vb, out);
}
